// head_6760278524422
// MI455X (gfx1250) — matrix operand order repaired and checked by reading, not compiled, not run on hardware
//
#include <hip/hip_runtime.h>
#include <hip/hip_bf16.h>
#include <stdint.h>

#define BATCH 1024
#define TSEQ  256
#define CEMB  384
#define HDIM  64

typedef __attribute__((ext_vector_type(16))) __bf16 v16bf;
typedef __attribute__((ext_vector_type(8)))  float  v8f;

static __device__ __forceinline__ __bf16 tobf(float f) { return (__bf16)f; }

union BFrag {           // one WMMA 16-element bf16 fragment = 32 bytes
  v16bf v;
  uint4 u[2];
};

// ---------------------------------------------------------------------------
// Kernel 0: weight prep.  Wk/Wq/Wv f32 [384][64]  ->  WT bf16 [3][64][384]
// (transposed) so projection B-fragments become contiguous b128 loads.
// ---------------------------------------------------------------------------
__global__ __launch_bounds__(256) void wprep_kernel(
    const float* __restrict__ Wk, const float* __restrict__ Wq,
    const float* __restrict__ Wv, __bf16* __restrict__ WT)
{
  const int idx = blockIdx.x * 256 + threadIdx.x;   // 0 .. 73727
  const int mat = idx / (HDIM * CEMB);
  const int rem = idx % (HDIM * CEMB);
  const int n   = rem / CEMB;
  const int c   = rem % CEMB;
  const float* W = (mat == 0) ? Wk : ((mat == 1) ? Wq : Wv);
  WT[idx] = tobf(W[c * HDIM + n]);
}

// ---------------------------------------------------------------------------
// Kernel 1: qkv projection.  One wave -> one 16x64 bf16 tile of {k,q,v}.
// K-loop fully unrolled, load phase split from the 4-WMMA compute phase so
// the scheduler overlaps next-step loads with current-step matrix math.
// ---------------------------------------------------------------------------
__global__ __launch_bounds__(256) void qkv_proj_kernel(
    const float* __restrict__ x, const __bf16* __restrict__ WT,
    __bf16* __restrict__ kbuf, __bf16* __restrict__ qbuf,
    __bf16* __restrict__ vT)
{
  const int lane = threadIdx.x & 31;
  const int wave = threadIdx.x >> 5;
  const int gw   = blockIdx.x * 8 + wave;
  const int mat  = gw % 3;            // 0=k, 1=q, 2=v
  const int rowTile = gw / 3;         // 0 .. 16383
  const int m0   = rowTile * 16;

  const int lrow = lane & 15;
  const int half = lane >> 4;

  v8f acc[4] = {};
  const float*  xrowBase = x + (size_t)(m0 + lrow) * CEMB;
  const __bf16* wtBase   = WT + (size_t)mat * HDIM * CEMB;

#pragma unroll
  for (int kk = 0; kk < CEMB; kk += 32) {
    // ---- load phase: A chunks + all 4 B fragments
    const float* xrow = xrowBase + kk;
    float4 xa = *(const float4*)(xrow + 8 * half);
    float4 xb = *(const float4*)(xrow + 8 * half + 4);
    float4 xc = *(const float4*)(xrow + 16 + 8 * half);
    float4 xd = *(const float4*)(xrow + 16 + 8 * half + 4);
    BFrag bfr[4];
#pragma unroll
    for (int nt = 0; nt < 4; ++nt) {
      const __bf16* wp =
          wtBase + (size_t)(nt * 16 + lrow) * CEMB + kk + 8 * half;
      bfr[nt].u[0] = *(const uint4*)(wp);
      bfr[nt].u[1] = *(const uint4*)(wp + 16);
    }
    // ---- compute phase: cvt + 4 WMMAs
    v16bf a;
    a[0]=tobf(xa.x); a[1]=tobf(xa.y); a[2]=tobf(xa.z); a[3]=tobf(xa.w);
    a[4]=tobf(xb.x); a[5]=tobf(xb.y); a[6]=tobf(xb.z); a[7]=tobf(xb.w);
    a[8]=tobf(xc.x); a[9]=tobf(xc.y); a[10]=tobf(xc.z); a[11]=tobf(xc.w);
    a[12]=tobf(xd.x); a[13]=tobf(xd.y); a[14]=tobf(xd.z); a[15]=tobf(xd.w);
#pragma unroll
    for (int nt = 0; nt < 4; ++nt)
      acc[nt] = __builtin_amdgcn_wmma_f32_16x16x32_bf16(
          false, a, false, bfr[nt].v, (short)0, acc[nt], false, false);
  }

  // --- store (C/D layout: row = r + 8*half, col = lane&15)
  if (mat == 2) {
    const int b = m0 >> 8;                 // whole tile in one batch
    const int tbase = (m0 & 255) + 8 * half;
#pragma unroll
    for (int nt = 0; nt < 4; ++nt) {
      const int col = nt * 16 + lrow;
      union { __bf16 h[8]; uint4 u; } pk;
#pragma unroll
      for (int r = 0; r < 8; ++r) pk.h[r] = tobf(acc[nt][r]);
      *(uint4*)&vT[((size_t)b * HDIM + col) * TSEQ + tbase] = pk.u;  // packed
    }
  } else {
    __bf16* dst = (mat == 0) ? kbuf : qbuf;
#pragma unroll
    for (int nt = 0; nt < 4; ++nt) {
      const int col = nt * 16 + lrow;
#pragma unroll
      for (int r = 0; r < 8; ++r)
        dst[(size_t)(m0 + r + 8 * half) * HDIM + col] = tobf(acc[nt][r]);
    }
  }
}

// ---------------------------------------------------------------------------
// Kernel 2: causal attention, flash style.  4 waves / block, one wave per
// 16-query-row tile.  Scores staged in wave-private LDS (16 KB / wave).
// Score loop software-pipelined (preload kt+1's k-fragments).
// ---------------------------------------------------------------------------
__global__ __launch_bounds__(128) void attn_kernel(
    const __bf16* __restrict__ qbuf, const __bf16* __restrict__ kbuf,
    const __bf16* __restrict__ vT, float* __restrict__ out)
{
  __shared__ float sbuf[4][16 * 256];   // 64 KB

  const int lane = threadIdx.x & 31;
  const int wave = threadIdx.x >> 5;
  const int b    = blockIdx.x >> 2;
  const int rt   = (blockIdx.x & 3) * 4 + wave;   // query row-tile 0..15
  const int t0   = rt * 16;
  const int lrow = lane & 15;
  const int half = lane >> 4;
  const float scale = 0.051031036307982884f;      // 384^-0.5

  float* S = &sbuf[wave][0];

  // --- q A-fragments (H dim K = 0..31 / 32..63): 16B-chunk vector loads
  BFrag aq0, aq1;
  {
    const __bf16* qrow = qbuf + ((size_t)b * TSEQ + t0 + lrow) * HDIM;
    aq0.u[0] = *(const uint4*)(qrow + 16 * half);
    aq0.u[1] = *(const uint4*)(qrow + 8 + 16 * half);
    aq1.u[0] = *(const uint4*)(qrow + 32 + 16 * half);
    aq1.u[1] = *(const uint4*)(qrow + 40 + 16 * half);
  }

  const int nkeys = t0 + 16;                // causal: keys 0 .. t0+15
  const int nk2   = (nkeys + 31) & ~31;     // pad to K-step of 32

  if (nk2 > nkeys) {                        // zero prob pad columns
#pragma unroll
    for (int j = 0; j < 8; ++j)
      S[lrow * 256 + nkeys + half * 8 + j] = 0.0f;
  }

  // --- score pass: S = scale * q . k^T (causally masked), pipelined
  const __bf16* kbase = kbuf + ((size_t)b * TSEQ + lrow) * HDIM + 16 * half;
  BFrag nb0, nb1;
  {
    const uint4* kp = (const uint4*)(kbase);
    nb0.u[0] = kp[0]; nb0.u[1] = kp[1];
    nb1.u[0] = kp[4]; nb1.u[1] = kp[5];
  }
  for (int kt = 0; kt <= rt; ++kt) {
    BFrag bk0 = nb0, bk1 = nb1;
    if (kt < rt) {                          // preload next key tile
      const uint4* kp = (const uint4*)(kbase + (size_t)(kt + 1) * 16 * HDIM);
      nb0.u[0] = kp[0]; nb0.u[1] = kp[1];
      nb1.u[0] = kp[4]; nb1.u[1] = kp[5];
    }
    v8f acc = {};
    acc = __builtin_amdgcn_wmma_f32_16x16x32_bf16(false, aq0.v, false, bk0.v,
                                                  (short)0, acc, false, false);
    acc = __builtin_amdgcn_wmma_f32_16x16x32_bf16(false, aq1.v, false, bk1.v,
                                                  (short)0, acc, false, false);
    const int ks = kt * 16;
    const int keycol = ks + lrow;
#pragma unroll
    for (int r = 0; r < 8; ++r) {
      const int m = t0 + r + 8 * half;
      float s = acc[r] * scale;
      if (keycol > m) s = -3.0e38f;          // causal mask
      S[(r + 8 * half) * 256 + keycol] = s;
    }
  }

  // --- softmax per row (2 lanes/row; combine via lane-xor-16)
  float mx = -3.0e38f;
  for (int c = half; c < nkeys; c += 2)
    mx = fmaxf(mx, S[lrow * 256 + c]);
  mx = fmaxf(mx, __shfl_xor(mx, 16, 32));
  float sum = 0.0f;
  for (int c = half; c < nkeys; c += 2) {
    const float e = __expf(S[lrow * 256 + c] - mx);
    S[lrow * 256 + c] = e;                   // unnormalized probs in place
    sum += e;
  }
  sum += __shfl_xor(sum, 16, 32);
  const float linv = 1.0f / sum;

  float linvRow[8];
#pragma unroll
  for (int r = 0; r < 8; ++r)
    linvRow[r] = __shfl(linv, r + 8 * half, 32);

  // --- output pass: acc_o[16x64] += P(bf16) . v ; load phase then 4 WMMAs
  v8f acco[4] = {};
  for (int sk = 0; sk < nk2; sk += 32) {
    const float* prow = S + lrow * 256 + sk;
    float4 pa = *(const float4*)(prow + 8 * half);        // ds_load_b128
    float4 pb = *(const float4*)(prow + 8 * half + 4);
    float4 pc = *(const float4*)(prow + 16 + 8 * half);
    float4 pd = *(const float4*)(prow + 16 + 8 * half + 4);
    BFrag bv[4];
#pragma unroll
    for (int nt = 0; nt < 4; ++nt) {
      const __bf16* vrow =
          vT + ((size_t)b * HDIM + nt * 16 + lrow) * TSEQ + sk + 8 * half;
      bv[nt].u[0] = *(const uint4*)(vrow);
      bv[nt].u[1] = *(const uint4*)(vrow + 16);
    }
    v16bf ap;
    ap[0]=tobf(pa.x); ap[1]=tobf(pa.y); ap[2]=tobf(pa.z); ap[3]=tobf(pa.w);
    ap[4]=tobf(pb.x); ap[5]=tobf(pb.y); ap[6]=tobf(pb.z); ap[7]=tobf(pb.w);
    ap[8]=tobf(pc.x); ap[9]=tobf(pc.y); ap[10]=tobf(pc.z); ap[11]=tobf(pc.w);
    ap[12]=tobf(pd.x); ap[13]=tobf(pd.y); ap[14]=tobf(pd.z); ap[15]=tobf(pd.w);
#pragma unroll
    for (int nt = 0; nt < 4; ++nt)
      acco[nt] = __builtin_amdgcn_wmma_f32_16x16x32_bf16(
          false, ap, false, bv[nt].v, (short)0, acco[nt], false, false);
  }

  // --- store fp32 output with folded 1/l normalization
#pragma unroll
  for (int nt = 0; nt < 4; ++nt) {
    const int col = nt * 16 + lrow;
#pragma unroll
    for (int r = 0; r < 8; ++r) {
      const int m = t0 + r + 8 * half;
      out[((size_t)b * TSEQ + m) * HDIM + col] = acco[nt][r] * linvRow[r];
    }
  }
}

// ---------------------------------------------------------------------------
extern "C" void kernel_launch(void* const* d_in, const int* in_sizes, int n_in,
                              void* d_out, int out_size, void* d_ws,
                              size_t ws_size, hipStream_t stream) {
  const float* x  = (const float*)d_in[0];
  const float* Wk = (const float*)d_in[1];
  const float* Wq = (const float*)d_in[2];
  const float* Wv = (const float*)d_in[3];
  float* out = (float*)d_out;

  // workspace (bf16): kbuf | qbuf | vT | WT   -> ~100.8 MB
  const size_t nProj = (size_t)BATCH * TSEQ * HDIM;  // 16,777,216 elems
  char* ws = (char*)d_ws;
  __bf16* kbuf = (__bf16*)(ws);
  __bf16* qbuf = (__bf16*)(ws + 2 * nProj);
  __bf16* vT   = (__bf16*)(ws + 4 * nProj);
  __bf16* WT   = (__bf16*)(ws + 6 * nProj);          // 3*64*384 bf16

  wprep_kernel<<<(3 * HDIM * CEMB) / 256, 256, 0, stream>>>(Wk, Wq, Wv, WT);
  // 16384 row tiles * 3 matrices = 49152 wave-tasks / 8 waves per block
  qkv_proj_kernel<<<6144, 256, 0, stream>>>(x, WT, kbuf, qbuf, vT);
  // 1024 batches * 4 tile-groups, 4 waves (row tiles) per block
  attn_kernel<<<4096, 128, 0, stream>>>(qbuf, kbuf, vT, out);
}
